// MHRetention_67336497267250
// MI455X (gfx1250) — matrix operand order repaired and checked by reading, not compiled, not run on hardware
//
#include <hip/hip_runtime.h>
#include <hip/hip_bf16.h>
#include <math.h>

typedef __bf16 bf16_t;
typedef bf16_t v16bf __attribute__((ext_vector_type(16)));
typedef bf16_t v8bf  __attribute__((ext_vector_type(8)));
typedef float  v8f   __attribute__((ext_vector_type(8)));

#define EMB   1024
#define HEADS 16
#define HD    64
#define BATCH 4
#define SEQ   4096
#define MTOT  (BATCH * SEQ)   // 16384 rows

// ---------------- fp32 -> bf16 convert ----------------
__global__ void cvt_f32_bf16(const float* __restrict__ in, bf16_t* __restrict__ out, int n) {
    int i = blockIdx.x * blockDim.x + threadIdx.x;
    for (; i < n; i += gridDim.x * blockDim.x) out[i] = (bf16_t)in[i];
}

// ---------------- fragment helpers ----------------
static __device__ inline v16bf pack16(v8bf lo, v8bf hi) {
    v16bf r;
#pragma unroll
    for (int i = 0; i < 8; ++i) { r[i] = lo[i]; r[i + 8] = hi[i]; }
    return r;
}

// A-matrix 16x32 bf16 fragment (ISA layout): lane<16 holds K {0..7,16..23},
// lane>=16 holds K {8..15,24..31} for row M=lane&15. rowPtr = row base (32 K elems).
static __device__ inline v16bf load_afrag(const bf16_t* rowPtr, int laneHi) {
    const bf16_t* p = rowPtr + laneHi * 8;
    return pack16(*(const v8bf*)p, *(const v8bf*)(p + 16));
}

// B-matrix 32x16 bf16 fragment: lane = column N, lane<16 holds K {0..7,16..23},
// lane>=16 holds K {8..15,24..31}. rowPtr = base of column n's K run (32 elems, N-major x K).
static __device__ inline v16bf load_bfrag(const bf16_t* rowPtr, int laneHi) {
    const bf16_t* p = rowPtr + laneHi * 8;
    return pack16(*(const v8bf*)p, *(const v8bf*)(p + 16));
}

static __device__ inline v8f wmma_bf16(v16bf a, v16bf b, v8f c) {
    return __builtin_amdgcn_wmma_f32_16x16x32_bf16(false, a, false, b, (short)0, c, false, false);
}

// ---------------- gfx1250 async global->LDS copy (ASYNCcnt path) ----------------
// VDST = per-lane LDS byte offset (addrspace(3) value = low 32 bits of flat ptr),
// VADDR = 64-bit global address, GV mode.  ISA §15.18 opcode 98.
static __device__ inline void async_ld_b128(void* ldsPtr, const void* gptr) {
    uint32_t lds = (uint32_t)(uintptr_t)ldsPtr;
    unsigned long long g = (unsigned long long)(uintptr_t)gptr;
    asm volatile("global_load_async_to_lds_b128 %0, %1, off"
                 :: "v"(lds), "v"(g) : "memory");
}
static __device__ inline void wait_async0() {
#if __has_builtin(__builtin_amdgcn_s_wait_asynccnt)
    __builtin_amdgcn_s_wait_asynccnt(0);
#else
    asm volatile("s_wait_asynccnt 0x0" ::: "memory");
#endif
}

// ---------------- tiled GEMM: C(MxN) = A(MxK,row) * W(NxK,row)^T ----------------
// EPI: 0 = store f32 (row layout)
//      1 = relu*0.125 -> bf16 (row layout)          [q]
//      2 = silu       -> bf16 (row layout)          [u]
//      3 = relu*0.125 -> bf16 transposed (b,e,l)    [k^T]
//      4 = plain      -> bf16 transposed (b,e,l)    [v^T]
#define BM 128
#define BN 128
#define BK 32
#define LSTR 40   // BK + 8 pad (keeps 16B alignment, breaks bank conflicts)

template <int EPI>
__global__ __launch_bounds__(256) void gemm_bf16(const bf16_t* __restrict__ A,
                                                 const bf16_t* __restrict__ W,
                                                 void* __restrict__ Cout,
                                                 int M, int N, int K) {
    __shared__ bf16_t lsA[BM * LSTR];
    __shared__ bf16_t lsB[BN * LSTR];

    const int tid = threadIdx.x;
    const int wave = tid >> 5, lane = tid & 31;
    const int laneHi = lane >> 4, lane16 = lane & 15;
    const int wm = wave & 3;   // 4 waves over M -> 32 rows each
    const int wn = wave >> 2;  // 2 waves over N -> 64 cols each
    const int m0 = blockIdx.x * BM;
    const int n0 = blockIdx.y * BN;

    v8f acc[2][4];
#pragma unroll
    for (int i = 0; i < 2; ++i)
#pragma unroll
        for (int j = 0; j < 4; ++j) acc[i][j] = (v8f)(0.0f);

    for (int k0 = 0; k0 < K; k0 += BK) {
        // stage A,B tiles via async global->LDS: 512 chunks of 16B per matrix
#pragma unroll
        for (int c = 0; c < 2; ++c) {
            int ch = tid + c * 256;
            int row = ch >> 2, cc = (ch & 3) * 8;
            const bf16_t* ga = &A[(size_t)(m0 + row) * K + k0 + cc];
            const bf16_t* gb = &W[(size_t)(n0 + row) * K + k0 + cc];
            async_ld_b128(&lsA[row * LSTR + cc], ga);
            async_ld_b128(&lsB[row * LSTR + cc], gb);
            if (k0 + BK < K) {  // pull next K-tile toward L2 (global_prefetch_b8)
                __builtin_prefetch(ga + BK, 0, 0);
                __builtin_prefetch(gb + BK, 0, 0);
            }
        }
        wait_async0();
        __syncthreads();

        v16bf af[2], bf[4];
#pragma unroll
        for (int i = 0; i < 2; ++i)
            af[i] = load_afrag(&lsA[(wm * 32 + i * 16 + lane16) * LSTR], laneHi);
#pragma unroll
        for (int j = 0; j < 4; ++j)
            bf[j] = load_bfrag(&lsB[(wn * 64 + j * 16 + lane16) * LSTR], laneHi);

#pragma unroll
        for (int i = 0; i < 2; ++i)
#pragma unroll
            for (int j = 0; j < 4; ++j)
                acc[i][j] = wmma_bf16(af[i], bf[j], acc[i][j]);
        __syncthreads();
    }

    // epilogue: C tile element e -> row = +e+(laneHi?8:0), col = lane16
    const int bb = m0 / SEQ;  // batch index (BM divides SEQ, blocks never straddle)
#pragma unroll
    for (int i = 0; i < 2; ++i)
#pragma unroll
        for (int j = 0; j < 4; ++j)
#pragma unroll
            for (int e = 0; e < 8; ++e) {
                size_t r = (size_t)(m0 + wm * 32 + i * 16 + e + laneHi * 8);
                size_t c = (size_t)(n0 + wn * 64 + j * 16 + lane16);
                float v = acc[i][j][e];
                if (EPI == 0) {
                    ((float*)Cout)[r * N + c] = v;
                } else if (EPI == 1 || EPI == 2) {
                    if (EPI == 1) v = fmaxf(v, 0.0f) * 0.125f;           // relu/sqrt(hd)
                    if (EPI == 2) v = v * (1.0f / (1.0f + __expf(-v)));  // silu
                    ((bf16_t*)Cout)[r * N + c] = (bf16_t)v;
                } else {
                    if (EPI == 3) v = fmaxf(v, 0.0f) * 0.125f;           // relu/sqrt(hd)
                    size_t l = r - (size_t)bb * SEQ;
                    ((bf16_t*)Cout)[((size_t)bb * EMB + c) * SEQ + l] = (bf16_t)v;
                }
            }
}

// ---------------- kv = K^T V per (b,h): (64 x 4096) @ (4096 x 64) ----------------
// kT/vT are stored (b, e, l): contraction dim l is contiguous -> direct global
// fragment loads, zero LDS, zero barriers. 8 waves x 2 tiles = 16 (4x4) tiles.
__global__ __launch_bounds__(256) void kv_kernel(const bf16_t* __restrict__ kT,
                                                 const bf16_t* __restrict__ vT,
                                                 float* __restrict__ kv) {
    const int bh = blockIdx.x;
    const int b = bh >> 4, h = bh & 15;
    const int tid = threadIdx.x;
    const int wave = tid >> 5, lane = tid & 31;
    const int laneHi = lane >> 4, lane16 = lane & 15;

    const int tm = wave >> 1;            // 0..3
    const int tn0 = (wave & 1) * 2;      // 0 or 2
    const bf16_t* kbase = kT + ((size_t)b * EMB + h * HD) * SEQ;
    const bf16_t* vbase = vT + ((size_t)b * EMB + h * HD) * SEQ;

    v8f acc[2];
    acc[0] = (v8f)(0.0f); acc[1] = (v8f)(0.0f);

    for (int l0 = 0; l0 < SEQ; l0 += 32) {
        v16bf a = load_afrag(&kbase[(size_t)(tm * 16 + lane16) * SEQ + l0], laneHi);
#pragma unroll
        for (int j = 0; j < 2; ++j) {
            v16bf bb = load_bfrag(&vbase[(size_t)((tn0 + j) * 16 + lane16) * SEQ + l0], laneHi);
            acc[j] = wmma_bf16(a, bb, acc[j]);
        }
    }

#pragma unroll
    for (int j = 0; j < 2; ++j)
#pragma unroll
        for (int e = 0; e < 8; ++e) {
            int row = tm * 16 + e + laneHi * 8;       // d_k
            int col = (tn0 + j) * 16 + lane16;        // d_v
            kv[(size_t)bh * (HD * HD) + row * HD + col] = acc[j][e];
        }
}

// ---------------- out = Q @ kv, SRMS-norm, * u -> bf16 ----------------
__global__ __launch_bounds__(256) void out_norm_kernel(const bf16_t* __restrict__ qb,
                                                       const float* __restrict__ kv,
                                                       const bf16_t* __restrict__ ub,
                                                       bf16_t* __restrict__ attn) {
    __shared__ bf16_t kvT[64 * 72];  // [n][k] = kv[k][n], padded rows

    const int bh = blockIdx.x;
    const int b = bh >> 4, h = bh & 15;
    const int mtile = blockIdx.y;
    const int tid = threadIdx.x;
    const int wave = tid >> 5, lane = tid & 31;
    const int laneHi = lane >> 4, lane16 = lane & 15;

    for (int i = tid; i < HD * HD; i += 256) {
        int k = i >> 6, n = i & 63;
        kvT[n * 72 + k] = (bf16_t)kv[(size_t)bh * (HD * HD) + i];
    }
    __syncthreads();

    const size_t arow = (size_t)b * SEQ + mtile * 128 + wave * 16 + lane16;  // A-frag row
    v8f acc[4];
#pragma unroll
    for (int j = 0; j < 4; ++j) acc[j] = (v8f)(0.0f);

#pragma unroll
    for (int k0 = 0; k0 < HD; k0 += 32) {
        const bf16_t* pa = &qb[arow * EMB + h * HD + k0];
        v16bf a = load_afrag(pa, laneHi);
#pragma unroll
        for (int tn = 0; tn < 4; ++tn) {
            v16bf bb = load_bfrag(&kvT[(tn * 16 + lane16) * 72 + k0], laneHi);
            acc[tn] = wmma_bf16(a, bb, acc[tn]);
        }
    }

    // SRMS norm: row of 64 values = 16 lanes (this half) x 4 accs, per element e.
    float inv[8];
#pragma unroll
    for (int e = 0; e < 8; ++e) {
        float ss = 0.0f;
#pragma unroll
        for (int tn = 0; tn < 4; ++tn) { float v = acc[tn][e]; ss += v * v; }
        // xor masks < 16 stay inside each 16-lane half (rows e vs e+8) — wave32 safe
        ss += __shfl_xor(ss, 1);
        ss += __shfl_xor(ss, 2);
        ss += __shfl_xor(ss, 4);
        ss += __shfl_xor(ss, 8);
        float nn = sqrtf(ss) * 0.125f;          // ||x * (1/sqrt(64))||
        inv[e] = 1.0f / fmaxf(nn, 1e-12f);
    }

#pragma unroll
    for (int tn = 0; tn < 4; ++tn)
#pragma unroll
        for (int e = 0; e < 8; ++e) {
            size_t row = (size_t)b * SEQ + mtile * 128 + wave * 16 + e + laneHi * 8;
            int col = h * HD + tn * 16 + lane16;
            float uval = (float)ub[row * EMB + col];
            attn[row * EMB + col] = (bf16_t)(acc[tn][e] * inv[e] * uval);
        }
}

// ---------------- launch ----------------
extern "C" void kernel_launch(void* const* d_in, const int* in_sizes, int n_in,
                              void* d_out, int out_size, void* d_ws, size_t ws_size,
                              hipStream_t stream) {
    const float* x  = (const float*)d_in[0];
    const float* Wq = (const float*)d_in[1];
    const float* Wk = (const float*)d_in[2];
    const float* Wv = (const float*)d_in[3];
    const float* Wu = (const float*)d_in[4];
    const float* Wo = (const float*)d_in[5];
    float* out = (float*)d_out;

    const size_t ME = (size_t)MTOT * EMB;   // 16,777,216 elems
    const size_t EE = (size_t)EMB * EMB;    //  1,048,576 elems

    bf16_t* base = (bf16_t*)d_ws;
    bf16_t* xb  = base;
    bf16_t* wqb = xb + ME;
    bf16_t* wkb = wqb + EE;
    bf16_t* wvb = wkb + EE;
    bf16_t* wub = wvb + EE;
    bf16_t* wob = wub + EE;
    bf16_t* qb  = wob + EE;
    bf16_t* kTb = qb + ME;    // (b, e, l) layout
    bf16_t* vTb = kTb + ME;   // (b, e, l) layout
    bf16_t* ub  = vTb + ME;
    float*  kvf = (float*)(ub + ME);
    bf16_t* attn = kTb;  // kTb dead after kv_kernel; reuse as attn buffer

    // converts
    cvt_f32_bf16<<<2048, 256, 0, stream>>>(x,  xb,  (int)ME);
    cvt_f32_bf16<<<1024, 256, 0, stream>>>(Wq, wqb, (int)EE);
    cvt_f32_bf16<<<1024, 256, 0, stream>>>(Wk, wkb, (int)EE);
    cvt_f32_bf16<<<1024, 256, 0, stream>>>(Wv, wvb, (int)EE);
    cvt_f32_bf16<<<1024, 256, 0, stream>>>(Wu, wub, (int)EE);
    cvt_f32_bf16<<<1024, 256, 0, stream>>>(Wo, wob, (int)EE);

    dim3 ggrid(MTOT / BM, EMB / BN);  // (128, 8)
    gemm_bf16<1><<<ggrid, 256, 0, stream>>>(xb, wqb, (void*)qb,  MTOT, EMB, EMB); // q
    gemm_bf16<3><<<ggrid, 256, 0, stream>>>(xb, wkb, (void*)kTb, MTOT, EMB, EMB); // k^T
    gemm_bf16<4><<<ggrid, 256, 0, stream>>>(xb, wvb, (void*)vTb, MTOT, EMB, EMB); // v^T
    gemm_bf16<2><<<ggrid, 256, 0, stream>>>(xb, wub, (void*)ub,  MTOT, EMB, EMB); // u=silu

    kv_kernel<<<BATCH * HEADS, 256, 0, stream>>>(kTb, vTb, kvf);

    dim3 ogrid(BATCH * HEADS, SEQ / 128);  // (64, 32)
    out_norm_kernel<<<ogrid, 256, 0, stream>>>(qb, kvf, ub, attn);

    gemm_bf16<0><<<ggrid, 256, 0, stream>>>(attn, wob, (void*)out, MTOT, EMB, EMB);
}
